// MBAB_69295002354088
// MI455X (gfx1250) — matrix operand order repaired and checked by reading, not compiled, not run on hardware
//
#include <hip/hip_runtime.h>
#include <hip/hip_bf16.h>

typedef _Float16 h16;
typedef __attribute__((ext_vector_type(16))) _Float16 v16h;
typedef __attribute__((ext_vector_type(8)))  float    v8f;

union V16U { v16h v; uint4 q[2]; };

#define HW    65536
#define WDIM  256
#define NB    4
#define NTOT  262144   /* NB*HW, == 2^18 */

__device__ __forceinline__ v8f wmma16x16x32(v16h a, v16h b, v8f c) {
  return __builtin_amdgcn_wmma_f32_16x16x32_f16(false, a, false, b, (short)0, c,
                                                false, false);
}

// ---------------------------------------------------------------- utilities
__global__ void k_f2h(const float* __restrict__ s, h16* __restrict__ d, int n) {
  int i = blockIdx.x * 256 + threadIdx.x;
  if (i < n) d[i] = (h16)s[i];
}
__global__ void k_zero(float* p, int n) {
  int i = blockIdx.x * 256 + threadIdx.x;
  if (i < n) p[i] = 0.0f;
}

// ------------------------------------------------------------------- GEMM
// out[o][n] = sum_k W[o][k] * IN[k][n]  (+bias). Activations are (B,Ctot,HW)
// f16 planes, routed over two channel segments. One block = one 16-row O-tile
// x 512 pixels (8 waves x 64 cols). W tile staged once to LDS (row-major);
// each K=32 chunk of the B panel staged COLUMN-MAJOR (transposed in regs at
// staging) so A and B fragments are each two contiguous ds_load_b128.
template<bool BIAS, bool STATS>
__global__ void __launch_bounds__(256)
k_gemm(const h16* __restrict__ W16, int K, int Otiles,
       const h16* __restrict__ s0, int s0ctot, int s0cbase, int ksplit,
       const h16* __restrict__ s1, int s1ctot, int s1cbase,
       const float* __restrict__ bias,
       h16* __restrict__ out, int octot, int obase,
       float* __restrict__ stats)
{
  constexpr int NP   = 512;   // pixels per block
  constexpr int KSTR = 40;    // padded per-pixel k stride (halves): 80B rows
  __shared__ __align__(16) h16 Bs[NP * KSTR];   // [n][k] column-major, 40 KB
  __shared__ __align__(16) h16 Ws[16 * 192];    // max K = 192

  const int tid  = threadIdx.x;
  const int ot   = blockIdx.x % Otiles;
  const int nb   = blockIdx.x / Otiles;
  const int n0   = nb * NP;
  const int bidx = n0 >> 16;
  const int p0   = n0 & (HW - 1);

  const int lane = tid & 31;
  const int l16  = lane & 15;
  const int hi   = lane >> 4;
  const int wave = tid >> 5;
  const int nc0  = wave * 64;
  const int kbA  = hi * 16;
  const int kbB  = hi * 16;

  // stage 16 x K weight tile (rows are contiguous in W16)
  {
    const h16* wsrc = W16 + (size_t)ot * 16 * K;
    for (int idx = tid; idx < 16 * K; idx += 256) Ws[idx] = wsrc[idx];
  }

  v8f zz = {0.f,0.f,0.f,0.f,0.f,0.f,0.f,0.f};
  v8f acc[4] = {zz, zz, zz, zz};

  for (int kc = 0; kc < K; kc += 32) {
    __syncthreads();   // protect Bs (and Ws on first pass)
    // stage 32 x 512 panel transposed: 16 row-pairs x 64 col-groups of 8
    for (int idx = tid; idx < 16 * 64; idx += 256) {
      const int rp   = idx >> 6;        // row pair 0..15  -> k = kc+2*rp
      const int col8 = idx & 63;        // 8-pixel group
      const int k  = kc + rp * 2;       // ksplit is even: pair never straddles
      const int kl = rp * 2;
      const h16* sp; int ct, cb, kk;
      if (k < ksplit) { sp = s0; ct = s0ctot; cb = s0cbase; kk = k; }
      else            { sp = s1; ct = s1ctot; cb = s1cbase; kk = k - ksplit; }
      const h16* gp = sp + (size_t)(bidx * ct + cb + kk) * HW + p0 + col8 * 8;
      uint4 q0 = *(const uint4*)gp;
      uint4 q1 = *(const uint4*)(gp + HW);
      if (k + 32 < ksplit) __builtin_prefetch(gp + (size_t)32 * HW, 0, 0);
      const unsigned short* ra = (const unsigned short*)&q0;
      const unsigned short* rb = (const unsigned short*)&q1;
      const int nbase = col8 * 8;
#pragma unroll
      for (int i = 0; i < 8; ++i) {
        unsigned int pk = (unsigned int)ra[i] | ((unsigned int)rb[i] << 16);
        *(unsigned int*)&Bs[(nbase + i) * KSTR + kl] = pk;
      }
    }
    __syncthreads();

    V16U a;
    {
      const h16* ap = &Ws[l16 * K + kc + kbA];
      a.q[0] = *(const uint4*)ap;
      a.q[1] = *(const uint4*)(ap + 8);
    }
#pragma unroll
    for (int s = 0; s < 4; ++s) {
      const int ncol = nc0 + s * 16 + l16;
      V16U b;
      const h16* bp = &Bs[ncol * KSTR + kbB];
      b.q[0] = *(const uint4*)bp;
      b.q[1] = *(const uint4*)(bp + 8);
      acc[s] = wmma16x16x32(a.v, b.v, acc[s]);
    }
  }

#pragma unroll
  for (int r = 0; r < 8; ++r) {
    const int ch = ot * 16 + r + hi * 8;
    const float bv = BIAS ? bias[ch] : 0.0f;
    float su = 0.0f, sq = 0.0f;
#pragma unroll
    for (int s = 0; s < 4; ++s) {
      float v = acc[s][r] + bv;
      out[(size_t)(bidx * octot + obase + ch) * HW + p0 + nc0 + s * 16 + l16] =
          (h16)v;
      su += v; sq += v * v;
    }
    if (STATS) {
#pragma unroll
      for (int m = 8; m >= 1; m >>= 1) {
        su += __shfl_xor(su, m, 16);
        sq += __shfl_xor(sq, m, 16);
      }
      if (l16 == 0) {
        atomicAdd(&stats[ch], su);
        atomicAdd(&stats[Otiles * 16 + ch], sq);
      }
    }
  }
}

// --------------------------------------------------- depthwise/grouped 3x3
// NIN inputs per group (1 or 2), 'SAME' zero padding, fused BN-stat atomics.
template<int NIN>
__global__ void __launch_bounds__(256)
k_dw3(const float* __restrict__ wt,
      const h16* __restrict__ s0, int s0ctot, int s0cbase,
      const h16* __restrict__ s1, int s1ctot, int s1cbase,
      h16* __restrict__ out, int octot, int obase,
      float* __restrict__ stats, int Ostats)
{
  const int g = blockIdx.y;
  const int n = blockIdx.x * 256 + threadIdx.x;
  const int b = n >> 16;
  const int p = n & (HW - 1);
  const int h = p >> 8;
  const int w = p & 255;

  float acc = 0.0f;
#pragma unroll
  for (int ic = 0; ic < NIN; ++ic) {
    const h16* base;
    if (NIN == 1) {
      base = s0 + (size_t)(b * s0ctot + s0cbase + g) * HW;
    } else {
      int cc = 2 * g + ic;
      if (cc < 48) base = s0 + (size_t)(b * s0ctot + s0cbase + cc) * HW;
      else         base = s1 + (size_t)(b * s1ctot + s1cbase + (cc - 48)) * HW;
    }
    const float* wg = wt + (size_t)(g * NIN + ic) * 9;
#pragma unroll
    for (int kh = 0; kh < 3; ++kh) {
      int hh = h + kh - 1;
      if ((unsigned)hh >= 256u) continue;
#pragma unroll
      for (int kw = 0; kw < 3; ++kw) {
        int w2 = w + kw - 1;
        if ((unsigned)w2 >= 256u) continue;
        acc += wg[kh * 3 + kw] * (float)base[hh * 256 + w2];
      }
    }
  }
  out[(size_t)(b * octot + obase + g) * HW + p] = (h16)acc;

  __shared__ float rs[256], rq[256];
  rs[threadIdx.x] = acc; rq[threadIdx.x] = acc * acc;
  __syncthreads();
  for (int st = 128; st > 0; st >>= 1) {
    if (threadIdx.x < st) {
      rs[threadIdx.x] += rs[threadIdx.x + st];
      rq[threadIdx.x] += rq[threadIdx.x + st];
    }
    __syncthreads();
  }
  if (threadIdx.x == 0) {
    atomicAdd(&stats[g], rs[0]);
    atomicAdd(&stats[Ostats + g], rq[0]);
  }
}

// ------------------------------------------------------------ BN (+ReLU)
__global__ void k_bnrelu(const h16* __restrict__ src, int sctot, int scbase,
                         h16* __restrict__ dst, int dctot, int dcbase,
                         const float* __restrict__ g, const float* __restrict__ bt,
                         const float* __restrict__ stats, int O, int relu)
{
  size_t i = (size_t)blockIdx.x * 256 + threadIdx.x;
  int c = (int)(i >> 18);
  int n = (int)(i & (NTOT - 1));
  int b = n >> 16, p = n & (HW - 1);
  float mean = stats[c] * (1.0f / NTOT);
  float var  = stats[O + c] * (1.0f / NTOT) - mean * mean;
  float sc = g[c] * rsqrtf(var + 1e-5f);
  float sh = bt[c] - mean * sc;
  float v = sc * (float)src[(size_t)(b * sctot + scbase + c) * HW + p] + sh;
  if (relu) v = fmaxf(v, 0.0f);
  dst[(size_t)(b * dctot + dcbase + c) * HW + p] = (h16)v;
}

// ------------------------- conv4 post: BN + ReLU + SE scale + ch shuffle
__global__ void k_fepost(const h16* __restrict__ src,
                         const float* __restrict__ g, const float* __restrict__ bt,
                         const float* __restrict__ stats,
                         const float* __restrict__ vse, h16* __restrict__ dst)
{
  size_t i = (size_t)blockIdx.x * 256 + threadIdx.x;
  int c = (int)(i >> 18);
  int n = (int)(i & (NTOT - 1));
  int b = n >> 16, p = n & (HW - 1);
  float mean = stats[c] * (1.0f / NTOT);
  float var  = stats[96 + c] * (1.0f / NTOT) - mean * mean;
  float sc = g[c] * rsqrtf(var + 1e-5f);
  float sh = bt[c] - mean * sc;
  float v = sc * (float)src[(size_t)(b * 96 + c) * HW + p] + sh;
  v = fmaxf(v, 0.0f) * vse[b * 96 + c];
  int dc = (c % 48) * 2 + (c / 48);      // channel shuffle
  dst[(size_t)(b * 96 + dc) * HW + p] = (h16)v;
}

// ------------------------------------------------------------- SE branch
__global__ void k_maxpool(const h16* __restrict__ src, float* __restrict__ vmax) {
  __shared__ float red[256];
  size_t base = (size_t)blockIdx.x * HW;
  float m = -1e30f;
  for (int i = threadIdx.x; i < HW; i += 256)
    m = fmaxf(m, (float)src[base + i]);
  red[threadIdx.x] = m;
  __syncthreads();
  for (int st = 128; st > 0; st >>= 1) {
    if (threadIdx.x < st)
      red[threadIdx.x] = fmaxf(red[threadIdx.x], red[threadIdx.x + st]);
    __syncthreads();
  }
  if (threadIdx.x == 0) vmax[blockIdx.x] = red[0];
}

__global__ void k_se(const float* __restrict__ vmax, const float* __restrict__ fc1,
                     const float* __restrict__ fc2, float* __restrict__ vse) {
  __shared__ float hid[40];
  int t = threadIdx.x;
  if (t < 40) {
    int b = t / 10, j = t % 10;
    float s = 0.0f;
    for (int k = 0; k < 96; ++k) s += fc1[j * 96 + k] * vmax[b * 96 + k];
    hid[t] = fmaxf(s, 0.0f);
  }
  __syncthreads();
  for (int idx = t; idx < 384; idx += 128) {
    int b = idx / 96, c = idx % 96;
    float s = 0.0f;
    for (int j = 0; j < 10; ++j) s += fc2[c * 10 + j] * hid[b * 10 + j];
    vse[idx] = 1.0f / (1.0f + __expf(-s));
  }
}

// --------------------------------------------------------- residual adds
__global__ void k_addres(const h16* __restrict__ fe, const float* __restrict__ x,
                         float* __restrict__ xr32, h16* __restrict__ xr16) {
  size_t i = (size_t)blockIdx.x * 256 + threadIdx.x;
  float v = (float)fe[i] + x[i];
  xr32[i] = v;
  xr16[i] = (h16)v;
}
__global__ void k_final(const h16* __restrict__ po, const float* __restrict__ xr32,
                        float* __restrict__ out) {
  size_t i = (size_t)blockIdx.x * 256 + threadIdx.x;
  out[i] = (float)po[i] + xr32[i];
}

// -------------------------------------------------- windowed self-attention
// One workgroup per window (shifted by WS/2 with wraparound on load & store).
// Qs/Ps rows padded to 40 halves, Vs stored transposed [c][p] so every WMMA
// fragment is two contiguous ds_load_b128. Pass1 = Gram tiles via WMMA +
// online softmax (shfl_xor width-16 row reductions); pass2 = recompute, P
// staged through LDS (D-layout -> A-layout), two WMMAs accumulate O = P @ V.
template<int WS>
__global__ void __launch_bounds__(32 * (WS * WS / 16))
k_attn(const h16* __restrict__ src, int sctot, int qc0,
       h16* __restrict__ dst, int dctot, int oc0)
{
  constexpr int P     = WS * WS;
  constexpr int NW    = P / 16;
  constexpr int PPAD  = (P < 32) ? 32 : P;
  constexpr int SH    = WS / 2;
  constexpr int NT    = P / 16;
  constexpr int NPAIR = PPAD / 32;
  constexpr int QSTR  = 40;        // Qs/Ps row stride (halves)
  constexpr int VSTR  = PPAD + 8;  // Vs row stride (halves)

  __shared__ __align__(16) h16 Qs[PPAD * QSTR];
  __shared__ __align__(16) h16 Vs[32 * VSTR];      // transposed: [c][p]
  __shared__ __align__(16) h16 Ps[NW * 16 * QSTR];

  const int nwx = WDIM / WS;
  const int wid = blockIdx.x;
  const int b   = wid / (nwx * nwx);
  const int wr  = wid % (nwx * nwx);
  const int wh  = wr / nwx;
  const int ww  = wr % nwx;

  for (int idx = threadIdx.x; idx < PPAD * 32; idx += blockDim.x) {
    int p = idx >> 5, c = idx & 31;
    h16 qv = (h16)0.f, vv = (h16)0.f;
    if (p < P) {
      int i = p / WS, j = p % WS;
      int hs = (wh * WS + i + SH) & 255;
      int cs = (ww * WS + j + SH) & 255;
      size_t o = (size_t)hs * 256 + cs;
      qv = src[(size_t)(b * sctot + qc0 + c) * HW + o];
      vv = src[(size_t)(b * sctot + qc0 + 32 + c) * HW + o];
    }
    Qs[p * QSTR + c] = qv;
    Vs[c * VSTR + p] = vv;
  }
  __syncthreads();

  const int wave = threadIdx.x >> 5;
  const int lane = threadIdx.x & 31;
  const int l16  = lane & 15;
  const int hi   = lane >> 4;
  const int row0 = wave * 16;
  const int kbA  = hi * 16;
  const int kbB  = hi * 16;

  V16U aq;
  {
    const h16* ap = &Qs[(row0 + l16) * QSTR + kbA];
    aq.q[0] = *(const uint4*)ap;
    aq.q[1] = *(const uint4*)(ap + 8);
  }

  float mx[8], sm[8];
#pragma unroll
  for (int r = 0; r < 8; ++r) { mx[r] = -1e30f; sm[r] = 0.0f; }

  v8f zz = {0.f,0.f,0.f,0.f,0.f,0.f,0.f,0.f};

  for (int ct = 0; ct < NT; ++ct) {
    V16U bq;
    {
      const h16* bp = &Qs[(ct * 16 + l16) * QSTR + kbB];
      bq.q[0] = *(const uint4*)bp;
      bq.q[1] = *(const uint4*)(bp + 8);
    }
    v8f S = wmma16x16x32(aq.v, bq.v, zz);
#pragma unroll
    for (int r = 0; r < 8; ++r) {
      float sv = S[r];
      float mt = sv;
#pragma unroll
      for (int m = 8; m >= 1; m >>= 1) mt = fmaxf(mt, __shfl_xor(mt, m, 16));
      float nm = fmaxf(mx[r], mt);
      float ex = __expf(sv - nm);
      float es = ex;
#pragma unroll
      for (int m = 8; m >= 1; m >>= 1) es += __shfl_xor(es, m, 16);
      sm[r] = sm[r] * __expf(mx[r] - nm) + es;
      mx[r] = nm;
    }
  }

  float inv[8];
#pragma unroll
  for (int r = 0; r < 8; ++r) inv[r] = 1.0f / sm[r];

  v8f o0 = zz, o1 = zz;
  h16* Pw = &Ps[wave * 16 * QSTR];
  for (int t = 0; t < NPAIR; ++t) {
#pragma unroll
    for (int hh = 0; hh < 2; ++hh) {
      int ct = 2 * t + hh;
      if (ct < NT) {
        V16U bq;
        const h16* bp = &Qs[(ct * 16 + l16) * QSTR + kbB];
        bq.q[0] = *(const uint4*)bp;
        bq.q[1] = *(const uint4*)(bp + 8);
        v8f S = wmma16x16x32(aq.v, bq.v, zz);
#pragma unroll
        for (int r = 0; r < 8; ++r)
          Pw[(r + hi * 8) * QSTR + hh * 16 + l16] =
              (h16)(__expf(S[r] - mx[r]) * inv[r]);
      } else {
#pragma unroll
        for (int r = 0; r < 8; ++r)
          Pw[(r + hi * 8) * QSTR + hh * 16 + l16] = (h16)0.f;
      }
    }
    V16U ap;
    {
      const h16* pp = &Pw[l16 * QSTR + kbA];
      ap.q[0] = *(const uint4*)pp;
      ap.q[1] = *(const uint4*)(pp + 8);
    }
    V16U bv0, bv1;
    {
      const h16* v0 = &Vs[l16 * VSTR + 32 * t + kbB];
      const h16* v1 = &Vs[(16 + l16) * VSTR + 32 * t + kbB];
      bv0.q[0] = *(const uint4*)v0;
      bv0.q[1] = *(const uint4*)(v0 + 8);
      bv1.q[0] = *(const uint4*)v1;
      bv1.q[1] = *(const uint4*)(v1 + 8);
    }
    o0 = wmma16x16x32(ap.v, bv0.v, o0);
    o1 = wmma16x16x32(ap.v, bv1.v, o1);
  }

#pragma unroll
  for (int r = 0; r < 8; ++r) {
    int pix = row0 + r + hi * 8;
    int i = pix / WS, j = pix % WS;
    int hs = (wh * WS + i + SH) & 255;
    int cs = (ww * WS + j + SH) & 255;
    size_t o = (size_t)hs * 256 + cs;
    dst[(size_t)(b * dctot + oc0 + l16) * HW + o]      = (h16)o0[r];
    dst[(size_t)(b * dctot + oc0 + 16 + l16) * HW + o] = (h16)o1[r];
  }
}

// ===========================================================================
extern "C" void kernel_launch(void* const* d_in, const int* in_sizes, int n_in,
                              void* d_out, int out_size, void* d_ws, size_t ws_size,
                              hipStream_t stream)
{
  (void)in_sizes; (void)n_in; (void)out_size; (void)ws_size;
  const float* x = (const float*)d_in[0];
  auto fe = [&](int s, int k) { return (const float*)d_in[1 + s * 17 + k]; };
  const float* pi_w  = (const float*)d_in[52];
  const float* pi_b  = (const float*)d_in[53];
  const float* pi_g  = (const float*)d_in[54];
  const float* pi_bt = (const float*)d_in[55];
  const float* t1_w  = (const float*)d_in[56];
  const float* t1_b  = (const float*)d_in[57];
  const float* t2_w  = (const float*)d_in[58];
  const float* t2_b  = (const float*)d_in[59];
  const float* po_w  = (const float*)d_in[60];
  const float* po_b  = (const float*)d_in[61];

  char* ws = (char*)d_ws;
  size_t off = 0;
  auto alloc = [&](size_t bytes) {
    size_t r = off;
    off += (bytes + 255) & ~(size_t)255;
    return r;
  };
  auto hb = [](int c) { return (size_t)c * NTOT * 2; };

  h16*   R0   = (h16*)(ws + alloc(hb(96)));    // x16 -> conv4tmp -> ycat
  h16*   feA  = (h16*)(ws + alloc(hb(96)));
  h16*   feB  = (h16*)(ws + alloc(hb(96)));    // -> po out
  h16*   x1   = (h16*)(ws + alloc(hb(192)));   // -> xp
  h16*   dwb  = (h16*)(ws + alloc(hb(64)));    // dw tmp -> tbuf
  h16*   cat  = (h16*)(ws + alloc(hb(144)));
  h16*   xr16 = (h16*)(ws + alloc(hb(96)));
  float* xr32 = (float*)(ws + alloc((size_t)96 * NTOT * 4));
  float* stats= (float*)(ws + alloc(384 * 4));
  float* vmax = (float*)(ws + alloc(384 * 4));
  float* vse  = (float*)(ws + alloc(384 * 4));
  h16*   wA   = (h16*)(ws + alloc(18432 * 2));
  h16*   wB   = (h16*)(ws + alloc(18432 * 2));
  h16*   wT1  = (h16*)(ws + alloc(6144 * 2));
  h16*   wT2  = (h16*)(ws + alloc(6144 * 2));
  h16*   wPo  = (h16*)(ws + alloc(9216 * 2));

  dim3 blk(256);
  auto f2h = [&](const float* s, h16* d, int n) {
    k_f2h<<<(n + 255) / 256, blk, 0, stream>>>(s, d, n);
  };
  auto zstats = [&]() { k_zero<<<2, blk, 0, stream>>>(stats, 384); };

  f2h(x, R0, 96 * NTOT);  // x -> f16

  const h16* stageIn[3]  = { R0, feA, feB };
  h16*       stageOut[3] = { feA, feB, feA };

  for (int s = 0; s < 3; ++s) {
    const h16* in = stageIn[s];
    f2h(fe(s, 0), wA, 192 * 96);
    f2h(fe(s, 12), wB, 96 * 192);
    k_maxpool<<<384, blk, 0, stream>>>(in, vmax);
    k_se<<<1, 128, 0, stream>>>(vmax, fe(s, 15), fe(s, 16), vse);

    zstats();
    k_gemm<false, true><<<12 * 512, blk, 0, stream>>>(
        wA, 96, 12, in, 96, 0, 96, in, 96, 0, nullptr, x1, 192, 0, stats);
    k_bnrelu<<<192 * NTOT / 256, blk, 0, stream>>>(
        x1, 192, 0, x1, 192, 0, fe(s, 1), fe(s, 2), stats, 192, 1);

    zstats();
    k_dw3<1><<<dim3(NTOT / 256, 48), blk, 0, stream>>>(
        fe(s, 3), x1, 192, 48, x1, 192, 48, dwb, 48, 0, stats, 48);
    k_bnrelu<<<48 * NTOT / 256, blk, 0, stream>>>(
        dwb, 48, 0, cat, 144, 0, fe(s, 4), fe(s, 5), stats, 48, 1);

    zstats();
    k_dw3<2><<<dim3(NTOT / 256, 48), blk, 0, stream>>>(
        fe(s, 6), cat, 144, 0, x1, 192, 96, dwb, 48, 0, stats, 48);
    k_bnrelu<<<48 * NTOT / 256, blk, 0, stream>>>(
        dwb, 48, 0, cat, 144, 48, fe(s, 7), fe(s, 8), stats, 48, 1);

    zstats();
    k_dw3<2><<<dim3(NTOT / 256, 48), blk, 0, stream>>>(
        fe(s, 9), cat, 144, 48, x1, 192, 144, dwb, 48, 0, stats, 48);
    k_bnrelu<<<48 * NTOT / 256, blk, 0, stream>>>(
        dwb, 48, 0, cat, 144, 96, fe(s, 10), fe(s, 11), stats, 48, 1);

    zstats();
    k_gemm<false, true><<<6 * 512, blk, 0, stream>>>(
        wB, 192, 6, x1, 192, 0, 48, cat, 144, 0, nullptr, R0, 96, 0, stats);
    k_fepost<<<96 * NTOT / 256, blk, 0, stream>>>(
        R0, fe(s, 13), fe(s, 14), stats, vse, stageOut[s]);
  }

  k_addres<<<96 * NTOT / 256, blk, 0, stream>>>(feA, x, xr32, xr16);

  f2h(pi_w, wA, 192 * 96);
  f2h(t1_w, wT1, 64 * 96);
  f2h(t2_w, wT2, 64 * 96);
  f2h(po_w, wPo, 96 * 96);

  zstats();
  k_gemm<true, true><<<12 * 512, blk, 0, stream>>>(
      wA, 96, 12, xr16, 96, 0, 96, xr16, 96, 0, pi_b, x1, 192, 0, stats);
  k_bnrelu<<<192 * NTOT / 256, blk, 0, stream>>>(
      x1, 192, 0, x1, 192, 0, pi_g, pi_bt, stats, 192, 0);

  k_attn<4><<<NB * 64 * 64, 32, 0, stream>>>(x1, 192, 0, R0, 96, 0);
  k_gemm<true, false><<<4 * 512, blk, 0, stream>>>(
      wT1, 96, 4, x1, 192, 64, 64, R0, 96, 0, t1_b, dwb, 64, 0, stats);
  k_attn<8><<<NB * 32 * 32, 128, 0, stream>>>(dwb, 64, 0, R0, 96, 32);
  k_gemm<true, false><<<4 * 512, blk, 0, stream>>>(
      wT2, 96, 4, x1, 192, 128, 64, R0, 96, 32, t2_b, dwb, 64, 0, stats);
  k_attn<16><<<NB * 16 * 16, 512, 0, stream>>>(dwb, 64, 0, R0, 96, 64);
  k_gemm<true, false><<<6 * 512, blk, 0, stream>>>(
      wPo, 96, 6, R0, 96, 0, 96, R0, 96, 0, po_b, feB, 96, 0, stats);
  k_final<<<96 * NTOT / 256, blk, 0, stream>>>(feB, xr32, (float*)d_out);
}
